// xLSTMStockForecaster_1460288881519
// MI455X (gfx1250) — matrix operand order repaired and checked by reading, not compiled, not run on hardware
//
#include <hip/hip_runtime.h>
#include <math.h>

// ---------------------------------------------------------------- types
typedef __attribute__((ext_vector_type(16))) __bf16 v16bf;
typedef __attribute__((ext_vector_type(8)))  __bf16 v8bf;
typedef __attribute__((ext_vector_type(4)))  __bf16 v4bf;
typedef __attribute__((ext_vector_type(8)))  float  f32x8;
typedef __attribute__((ext_vector_type(4)))  float  f32x4;

#define WMMA_BF16(A_, B_, C_) \
  __builtin_amdgcn_wmma_f32_16x16x32_bf16(false, (A_), false, (B_), (short)0, (C_), false, false)

__device__ inline unsigned short f2bf_raw(float f) {
  unsigned int u = __builtin_bit_cast(unsigned int, f);
  unsigned int r = u + 0x7fffu + ((u >> 16) & 1u);
  return (unsigned short)(r >> 16);
}
__device__ inline __bf16 bf_from_raw(unsigned short u) { return __builtin_bit_cast(__bf16, u); }
__device__ inline __bf16 f2bf(float f) { return bf_from_raw(f2bf_raw(f)); }
__device__ inline f32x8 fzero8() {
  f32x8 z;
  for (int i = 0; i < 8; ++i) z[i] = 0.f;
  return z;
}
// Swizzle within a 32-wide K block: swap K[8..15] <-> K[16..23]. After this
// permutation the ISA's interleaved 16-bit A-fragment layout is CONTIGUOUS
// per (lane, half): fragment = slots [half*16, half*16+16). 8-block granular,
// so 4-aligned runs remain contiguous after swizzling.
__device__ inline int swzK(int c) {
  return (c & ~24) | ((c & 8) << 1) | ((c & 16) >> 1);
}
__device__ inline float sigm(float x) { return 1.f / (1.f + expf(-x)); }
__device__ inline float gelu_t(float g) {
  return 0.5f * g * (1.f + tanhf(0.7978845608028654f * (g + 0.044715f * g * g * g)));
}

// block-wide {sum, sumsq} reduction (wave32). red0/red1: >= nwaves floats.
__device__ inline void blockred2(float& sum, float& sq, float* red0, float* red1) {
  #pragma unroll
  for (int off = 16; off; off >>= 1) { sum += __shfl_xor(sum, off); sq += __shfl_xor(sq, off); }
  int wid = threadIdx.x >> 5, nw = blockDim.x >> 5;
  if ((threadIdx.x & 31) == 0) { red0[wid] = sum; red1[wid] = sq; }
  __syncthreads();
  if (threadIdx.x == 0) {
    float a = 0.f, c = 0.f;
    for (int i = 0; i < nw; ++i) { a += red0[i]; c += red1[i]; }
    red0[0] = a; red1[0] = c;
  }
  __syncthreads();
  sum = red0[0]; sq = red1[0];
  __syncthreads();
}

// ---------------------------------------------------------------- generic bf16-WMMA GEMM
// C[M,N] = alpha * A[M,K] @ W[K,N] (+bias) (+= old C). Optional batching via blockIdx.z.
#define BM 64
#define BN 64
#define BKK 32
__global__ __launch_bounds__(128) void k_gemm(
    const float* __restrict__ A, const float* __restrict__ W,
    const float* __restrict__ bias, float* __restrict__ C,
    int M, int N, int K, int lda, int ldw, int ldc,
    long long aBatch, long long wBatch, long long cBatch,
    float alpha, int accumulate)
{
  __shared__ __bf16 As[BM][BKK];      // unswizzled [row][k]; row = 64B aligned
  __shared__ __bf16 BsT[BN][BKK];     // transposed: [col][k]; row = 64B aligned
  const int bz = blockIdx.z;
  A += (long long)bz * aBatch; W += (long long)bz * wBatch; C += (long long)bz * cBatch;
  const int row0 = blockIdx.y * BM, col0 = blockIdx.x * BN;
  const int tid = threadIdx.x, wid = tid >> 5, lane = tid & 31;
  const int lrow = lane & 15, half = lane >> 4;
  const bool fullM = (row0 + BM <= M);
  const bool fullN = (col0 + BN <= N);

  f32x8 acc[4];
  #pragma unroll
  for (int n = 0; n < 4; ++n) acc[n] = fzero8();

  for (int k0 = 0; k0 < K; k0 += BKK) {
    const bool fullK = (k0 + BKK <= K);
    if (fullM && fullK) {
      // fast path: float4 global loads, packed 8B LDS stores
      #pragma unroll
      for (int it = 0; it < 4; ++it) {
        int i = tid + it * 128;          // float4 index 0..511
        int r = i >> 3;                  // 8 float4 per 32-wide row
        int c4 = (i & 7) * 4;
        f32x4 vv = *(const f32x4*)(A + (long long)(row0 + r) * lda + k0 + c4);
        v4bf pk;
        pk[0] = f2bf(vv[0]); pk[1] = f2bf(vv[1]);
        pk[2] = f2bf(vv[2]); pk[3] = f2bf(vv[3]);
        *(v4bf*)&As[r][c4] = pk;
      }
    } else {
      for (int i = tid; i < BM * BKK; i += 128) {
        int r = i >> 5, c = i & 31;
        int gr = row0 + r, gc = k0 + c;
        float v = (gr < M && gc < K) ? A[(long long)gr * lda + gc] : 0.f;
        As[r][c] = f2bf(v);
      }
    }
    if (fullN && fullK) {
      #pragma unroll
      for (int it = 0; it < 4; ++it) {
        int i = tid + it * 128;          // float4 index 0..511
        int r = i >> 4;                  // 16 float4 per 64-wide row
        int c4 = (i & 15) * 4;
        f32x4 vv = *(const f32x4*)(W + (long long)(k0 + r) * ldw + col0 + c4);
        BsT[c4 + 0][r] = f2bf(vv[0]);
        BsT[c4 + 1][r] = f2bf(vv[1]);
        BsT[c4 + 2][r] = f2bf(vv[2]);
        BsT[c4 + 3][r] = f2bf(vv[3]);
      }
    } else {
      for (int i = tid; i < BKK * BN; i += 128) {
        int r = i >> 6, c = i & 63;
        int gr = k0 + r, gc = col0 + c;
        float v = (gr < K && gc < N) ? W[(long long)gr * ldw + gc] : 0.f;
        BsT[c][r] = f2bf(v);
      }
    }
    __syncthreads();
    const int ar = wid * 16 + lrow;
    v16bf af = *(const v16bf*)&As[ar][half * 16];      // 2x ds_load_b128
    #pragma unroll
    for (int n = 0; n < 4; ++n) {
      v16bf bfrag = *(const v16bf*)&BsT[n * 16 + lrow][half * 16];
      acc[n] = WMMA_BF16(af, bfrag, acc[n]);
    }
    __syncthreads();
  }
  #pragma unroll
  for (int n = 0; n < 4; ++n)
    #pragma unroll
    for (int r = 0; r < 8; ++r) {
      int grow = row0 + wid * 16 + r + 8 * half;
      int gcol = col0 + n * 16 + lrow;
      if (grow < M && gcol < N) {
        float v = acc[n][r] * alpha;
        if (bias) v += bias[gcol];
        long long o = (long long)grow * ldc + gcol;
        if (accumulate) v += C[o];
        C[o] = v;
      }
    }
}

// ---------------------------------------------------------------- input proj + layernorm
__global__ __launch_bounds__(256) void k_input_proj(
    const float* __restrict__ x, const float* __restrict__ w, const float* __restrict__ b,
    const float* __restrict__ lns, const float* __restrict__ lnb, float* __restrict__ h)
{
  long long row = blockIdx.x;
  __shared__ float tmp[512];
  __shared__ float red0[8], red1[8];
  const float* xr = x + row * 8;
  for (int d = threadIdx.x; d < 512; d += 256) {
    float a = b[d];
    #pragma unroll
    for (int f = 0; f < 8; ++f) a += xr[f] * w[f * 512 + d];
    tmp[d] = a;
  }
  __syncthreads();
  float sum = 0.f, sq = 0.f;
  for (int d = threadIdx.x; d < 512; d += 256) { float v = tmp[d]; sum += v; sq += v * v; }
  blockred2(sum, sq, red0, red1);
  float mu = sum / 512.f, var = sq / 512.f - mu * mu, inv = rsqrtf(var + 1e-5f);
  for (int d = threadIdx.x; d < 512; d += 256)
    h[row * 512 + d] = (tmp[d] - mu) * inv * lns[d] + lnb[d];
}

// ---------------------------------------------------------------- layernorm (row at x+row*ldrow)
__global__ __launch_bounds__(256) void k_layernorm(
    const float* __restrict__ x, const float* __restrict__ s, const float* __restrict__ b,
    float* __restrict__ out, int D, long long ldrow)
{
  long long row = blockIdx.x;
  const float* xr = x + row * ldrow;
  __shared__ float red0[8], red1[8];
  float sum = 0.f, sq = 0.f;
  for (int d = threadIdx.x; d < D; d += blockDim.x) { float v = xr[d]; sum += v; sq += v * v; }
  blockred2(sum, sq, red0, red1);
  float mu = sum / D, var = sq / D - mu * mu, inv = rsqrtf(var + 1e-5f);
  for (int d = threadIdx.x; d < D; d += blockDim.x)
    out[row * (long long)D + d] = (xr[d] - mu) * inv * s[d] + b[d];
}

// ---------------------------------------------------------------- causal conv (K=4) + silu
__global__ void k_conv_silu(const float* __restrict__ xin, const float* __restrict__ w,
                            const float* __restrict__ bias, float* __restrict__ out,
                            int T, int D, int ldx, long long total)
{
  long long idx = (long long)blockIdx.x * blockDim.x + threadIdx.x;
  if (idx >= total) return;
  int d = (int)(idx % D);
  long long bt = idx / D;
  int t = (int)(bt % T);
  long long b = bt / T;
  float a = bias[d];
  #pragma unroll
  for (int kk = 0; kk < 4; ++kk) {
    int ts = t - 3 + kk;
    if (ts >= 0) a += xin[(b * T + ts) * (long long)ldx + d] * w[kk * D + d];
  }
  out[bt * D + d] = a * sigm(a);
}

// ---------------------------------------------------------------- R pre-transpose to bf16
// RT[g][h][col][k] = R[g][h][k][col], bf16. 4*8*64*64 elements.
__global__ void k_prep_rt(const float* __restrict__ R, __bf16* __restrict__ RT)
{
  int idx = blockIdx.x * 256 + threadIdx.x;
  if (idx >= 4 * 8 * 64 * 64) return;
  int k = idx & 63;
  int col = (idx >> 6) & 63;
  int gh = idx >> 12;
  RT[((long long)gh * 64 + col) * 64 + k] = f2bf(R[((long long)gh * 64 + k) * 64 + col]);
}

// ---------------------------------------------------------------- sLSTM scan (WMMA recurrence)
// grid = 2 blocks of 16 batch rows; block = 512 threads (16 waves).
__global__ __launch_bounds__(512) void k_slstm_scan(
    const float* __restrict__ pre_i, const float* __restrict__ pre_f,
    const float* __restrict__ pre_z, const float* __restrict__ pre_o,
    const __bf16* __restrict__ RT,  // (4,8,64,64) [g][h][col][k] bf16
    float* __restrict__ hs, int T)
{
  __shared__ __bf16 hbf[16][544];    // unswizzled [row][d]; row 1088B (32B-mult)
  __shared__ float recf[16][512];
  const int tid = threadIdx.x;
  const int wid = tid >> 5, lane = tid & 31;
  const int lrow = lane & 15, half = lane >> 4;
  const int bbase = blockIdx.x * 16;
  const int d = tid;                 // owned column 0..511
  const int dsw = d;
  const int tA = wid * 2;
  const int head = tA >> 2;          // each wave: one head, two 16-col chunks
  const int nc0 = tA & 3, nc1 = (tA + 1) & 3;

  float cS[16], nS[16], mS[16], itv[16], igv[16];
  #pragma unroll
  for (int r = 0; r < 16; ++r) { cS[r] = 0.f; nS[r] = 0.f; mS[r] = 0.f; }
  for (int i = tid; i < 16 * 544; i += 512) hbf[i / 544][i % 544] = bf_from_raw((unsigned short)0);
  __syncthreads();

  for (int t = 0; t < T; ++t) {
    // A fragments (H tile 16x64 for this wave's head): contiguous k = 16*half + s, same order as B
    v16bf a0 = *(const v16bf*)&hbf[lrow][head * 64 + half * 16];
    v16bf a1 = *(const v16bf*)&hbf[lrow][head * 64 + 32 + half * 16];
    #pragma unroll 1
    for (int g = 0; g < 4; ++g) {
      #pragma unroll
      for (int ti = 0; ti < 2; ++ti) {
        const int nc = ti ? nc1 : nc0;
        const int col = nc * 16 + lrow;
        const __bf16* Rt = RT + (((long long)(g * 8 + head) * 64 + col) * 64);
        v16bf b0 = *(const v16bf*)(Rt + 16 * half);
        v16bf b1 = *(const v16bf*)(Rt + 32 + 16 * half);
        f32x8 dacc = fzero8();
        dacc = WMMA_BF16(a0, b0, dacc);
        dacc = WMMA_BF16(a1, b1, dacc);
        #pragma unroll
        for (int r = 0; r < 8; ++r)
          recf[r + 8 * half][head * 64 + nc * 16 + lrow] = dacc[r];
      }
      __syncthreads();
      if (g == 0) {
        #pragma unroll
        for (int r = 0; r < 16; ++r)
          itv[r] = pre_i[((long long)(bbase + r) * T + t) * 512 + d] + recf[r][d];
      } else if (g == 1) {
        #pragma unroll
        for (int r = 0; r < 16; ++r) {
          float ft = pre_f[((long long)(bbase + r) * T + t) * 512 + d] + recf[r][d];
          float mnew = fmaxf(ft + mS[r], itv[r]);
          igv[r] = expf(itv[r] - mnew);
          float fg = expf(ft + mS[r] - mnew);
          mS[r] = mnew;
          cS[r] = fg * cS[r];
          nS[r] = fg * nS[r] + igv[r];
        }
      } else if (g == 2) {
        #pragma unroll
        for (int r = 0; r < 16; ++r) {
          float zt = tanhf(pre_z[((long long)(bbase + r) * T + t) * 512 + d] + recf[r][d]);
          cS[r] += igv[r] * zt;
        }
      } else {
        #pragma unroll
        for (int r = 0; r < 16; ++r) {
          float ot = sigm(pre_o[((long long)(bbase + r) * T + t) * 512 + d] + recf[r][d]);
          float hnew = ot * cS[r] / fmaxf(nS[r], 1e-6f);
          hs[((long long)(bbase + r) * T + t) * 512 + d] = hnew;
          hbf[r][dsw] = f2bf(hnew);
        }
      }
      __syncthreads();
    }
  }
}

// ---------------------------------------------------------------- head-norm + residual (sLSTM)
__global__ __launch_bounds__(256) void k_headnorm_add(
    const float* __restrict__ hsb, const float* __restrict__ gn,
    float* __restrict__ h, int D, int dh)
{
  long long row = blockIdx.x;
  int wid = threadIdx.x >> 5, lane = threadIdx.x & 31;
  const float* xr = hsb + row * D + wid * dh;
  float sum = 0.f, sq = 0.f;
  for (int e = lane; e < dh; e += 32) { float v = xr[e]; sum += v; sq += v * v; }
  #pragma unroll
  for (int off = 16; off; off >>= 1) { sum += __shfl_xor(sum, off); sq += __shfl_xor(sq, off); }
  float mu = sum / dh, var = sq / dh - mu * mu, inv = rsqrtf(var + 1e-5f);
  for (int e = lane; e < dh; e += 32) {
    int dc = wid * dh + e;
    h[row * D + dc] += gn[dc] * (xr[e] - mu) * inv;
  }
}

// ---------------------------------------------------------------- mLSTM: head-norm + skip + *silu(z)
__global__ __launch_bounds__(256) void k_mlstm_fuse(
    const float* __restrict__ attn, const float* __restrict__ gn,
    const float* __restrict__ skip, const float* __restrict__ xc,
    const float* __restrict__ ud, float* __restrict__ fused,
    int DI, int dh, int ldud)
{
  long long row = blockIdx.x;
  int wid = threadIdx.x >> 5, lane = threadIdx.x & 31;
  const float* xr = attn + row * DI + wid * dh;
  float sum = 0.f, sq = 0.f;
  for (int e = lane; e < dh; e += 32) { float v = xr[e]; sum += v; sq += v * v; }
  #pragma unroll
  for (int off = 16; off; off >>= 1) { sum += __shfl_xor(sum, off); sq += __shfl_xor(sq, off); }
  float mu = sum / dh, var = sq / dh - mu * mu, inv = rsqrtf(var + 1e-5f);
  for (int e = lane; e < dh; e += 32) {
    int dc = wid * dh + e;
    float hn = (xr[e] - mu) * inv * gn[dc] + skip[dc] * xc[row * DI + dc];
    float z = ud[row * (long long)ldud + DI + dc];
    fused[row * DI + dc] = hn * (z * sigm(z));
  }
}

// ---------------------------------------------------------------- FF gelu(g)*up
__global__ void k_gelu_mul(const float* __restrict__ u, float* __restrict__ out,
                           int FFd, int ldu, int ldo, long long total)
{
  long long idx = (long long)blockIdx.x * blockDim.x + threadIdx.x;
  if (idx >= total) return;
  int j = (int)(idx % FFd);
  long long row = idx / FFd;
  float g = u[row * ldu + j], up = u[row * ldu + FFd + j];
  out[row * ldo + j] = gelu_t(g) * up;
}

// ---------------------------------------------------------------- mLSTM small gate GEMM (N=16)
__global__ void k_gates(const float* __restrict__ xc, const float* __restrict__ wif,
                        const float* __restrict__ bif, float* __restrict__ gates,
                        int DI, long long total)
{
  long long idx = (long long)blockIdx.x * blockDim.x + threadIdx.x;
  if (idx >= total) return;
  int j = (int)(idx & 15);
  long long row = idx >> 4;
  float acc = bif[j];
  const float* xr = xc + row * DI;
  for (int kk = 0; kk < DI; ++kk) acc += xr[kk] * wif[kk * 16 + j];
  gates[row * 16 + j] = acc;
}

// ---------------------------------------------------------------- gate log-sigmoid cumsum (per b,h)
__global__ void k_gatescan(const float* __restrict__ gates, float* __restrict__ cf,
                           float* __restrict__ logi, int T)
{
  int tid = threadIdx.x;            // 256 = 32*8
  int b = tid >> 3, h = tid & 7;
  float c = 0.f;
  for (int t = 0; t < T; ++t) {
    long long r = ((long long)b * T + t) * 16;
    float gi = gates[r + h];
    float gf = gates[r + 8 + h];
    float lf = fminf(gf, 0.f) - log1pf(expf(-fabsf(gf)));
    c += lf;
    cf[((long long)b * 8 + h) * T + t] = c;
    logi[((long long)b * 8 + h) * T + t] = gi;
  }
}

// ---------------------------------------------------------------- mLSTM attention (flash-style, WMMA)
// grid (qtiles, NH, B), block 32 (one wave). dh = 128.
__global__ __launch_bounds__(32) void k_mlstm_attn(
    const float* __restrict__ q, const float* __restrict__ k, const float* __restrict__ v,
    const float* __restrict__ cf, const float* __restrict__ logi,
    float* __restrict__ out, int T, int DI)
{
  __shared__ __bf16 VtT[128][16];    // transposed V tile: [col][row]; row 32B
  __shared__ __bf16 Pt[16][24];      // row 48B (16B-mult)
  __shared__ float cfq[16];
  const int lane = threadIdx.x;
  const int lrow = lane & 15, half = lane >> 4;
  const int qt = blockIdx.x, hh = blockIdx.y, b = blockIdx.z;
  const int t0 = qt * 16;
  const long long hOff = (long long)hh * 128;
  const float* cfh = cf + ((long long)b * 8 + hh) * T;
  const float* lih = logi + ((long long)b * 8 + hh) * T;

  v16bf zb;
  #pragma unroll
  for (int e = 0; e < 16; ++e) zb[e] = bf_from_raw((unsigned short)0);

  // Q fragments (16x128 -> 4 chunks of 16x32); two contiguous 8-float spans each
  v16bf aq[4];
  {
    int row = t0 + lrow;
    bool rv = row < T;
    const float* qr = q + ((long long)b * T + (rv ? row : 0)) * DI + hOff;
    #pragma unroll
    for (int cck = 0; cck < 4; ++cck) {
      const float* s1 = qr + cck * 32 + 16 * half;
      const float* s2 = qr + cck * 32 + 16 * half + 8;
      #pragma unroll
      for (int m = 0; m < 8; ++m) {
        aq[cck][m]     = f2bf(rv ? s1[m] : 0.f);
        aq[cck][8 + m] = f2bf(rv ? s2[m] : 0.f);
      }
    }
  }
  if (lane < 16) cfq[lane] = (t0 + lane < T) ? cfh[t0 + lane] : 0.f;
  __syncthreads();
  float cfq_l[8];
  #pragma unroll
  for (int r = 0; r < 8; ++r) cfq_l[r] = cfq[r + 8 * half];

  float mrun[8], srun[8];
  f32x8 acc[8];
  #pragma unroll
  for (int r = 0; r < 8; ++r) { mrun[r] = -__builtin_inff(); srun[r] = 0.f; }
  #pragma unroll
  for (int nn = 0; nn < 8; ++nn) acc[nn] = fzero8();

  for (int kt = 0; kt <= qt; ++kt) {
    const int s0 = kt * 16;
    // S = (Q) @ (K^T) for this 16x16 tile
    f32x8 S = fzero8();
    {
      int col = s0 + lrow;
      bool cv = col < T;
      const float* kr = k + ((long long)b * T + (cv ? col : 0)) * DI + hOff;
      #pragma unroll
      for (int cck = 0; cck < 4; ++cck) {
        v16bf bk;
        #pragma unroll
        for (int e = 0; e < 16; ++e)
          bk[e] = f2bf(cv ? kr[cck * 32 + 16 * half + e] : 0.f);
        S = WMMA_BF16(aq[cck], bk, S);
      }
    }
    // stage V tile (16 x 128) transposed into LDS as bf16
    for (int i = lane; i < 16 * 128; i += 32) {
      int r = i >> 7, cc = i & 127;
      int srow = s0 + r;
      float vv = (srow < T) ? v[((long long)b * T + srow) * DI + hOff + cc] : 0.f;
      VtT[cc][r] = f2bf(vv);
    }
    // gating + online rescale
    const int col = s0 + lrow;
    const float cfs = (col < T) ? cfh[col] : 0.f;
    const float lis = (col < T) ? lih[col] : 0.f;
    float p[8], scl[8];
    #pragma unroll
    for (int r = 0; r < 8; ++r) {
      int row = t0 + r + 8 * half;
      bool valid = (col <= row) && (col < T) && (row < T);
      float dm = valid ? (cfq_l[r] - cfs + lis) : -__builtin_inff();
      float rmax = dm;
      #pragma unroll
      for (int off = 8; off; off >>= 1) rmax = fmaxf(rmax, __shfl_xor(rmax, off));
      float mnew = fmaxf(mrun[r], rmax);
      float pe = 0.f, scale = 0.f;
      if (mnew != -__builtin_inff()) {
        pe = valid ? expf(dm - mnew) : 0.f;
        scale = expf(mrun[r] - mnew);
      }
      p[r] = pe * S[r];
      float rs = p[r];
      #pragma unroll
      for (int off = 8; off; off >>= 1) rs += __shfl_xor(rs, off);
      srun[r] = srun[r] * scale + rs;
      mrun[r] = mnew;
      scl[r] = scale;
    }
    #pragma unroll
    for (int nn = 0; nn < 8; ++nn)
      #pragma unroll
      for (int r = 0; r < 8; ++r) acc[nn][r] *= scl[r];
    // stage P (16x16 bf16) for the S@V WMMA (K padded to 32 with zeros)
    #pragma unroll
    for (int r = 0; r < 8; ++r) Pt[r + 8 * half][lrow] = f2bf(p[r]);
    __syncthreads();
    v16bf ap;
    {
      v8bf pl = *(const v8bf*)&Pt[lrow][0];    // contiguous 16B
      v8bf ph = *(const v8bf*)&Pt[lrow][8];
      #pragma unroll
      for (int m = 0; m < 8; ++m) {
        ap[m]     = half ? bf_from_raw((unsigned short)0) : pl[m];
        ap[8 + m] = half ? bf_from_raw((unsigned short)0) : ph[m];
      }
    }
    #pragma unroll
    for (int nn = 0; nn < 8; ++nn) {
      v16bf bv;
      if (half == 0) bv = *(const v16bf*)&VtT[nn * 16 + lrow][0];  // 2x ds_load_b128
      else           bv = zb;                                       // K>=16 rows are zero-padded
      acc[nn] = WMMA_BF16(ap, bv, acc[nn]);
    }
    __syncthreads();
  }
  // epilogue: norm = max(|rowsum|, exp(-m)), write out
  #pragma unroll
  for (int r = 0; r < 8; ++r) {
    int row = t0 + r + 8 * half;
    if (row >= T) continue;
    float nrm = fmaxf(fabsf(srun[r]), expf(-mrun[r]));
    float inv = 1.f / fmaxf(nrm, 1e-30f);
    #pragma unroll
    for (int nn = 0; nn < 8; ++nn)
      out[((long long)b * T + row) * DI + hOff + nn * 16 + lrow] = acc[nn][r] * inv;
  }
}

// ---------------------------------------------------------------- final per-head MLPs
__global__ __launch_bounds__(128) void k_heads(
    const float* __restrict__ last,
    const float* __restrict__ hw1, const float* __restrict__ hb1,
    const float* __restrict__ hw2, const float* __restrict__ hb2,
    const float* __restrict__ hw3, const float* __restrict__ hb3,
    float* __restrict__ out)
{
  const int hh = blockIdx.x, b = blockIdx.y;
  __shared__ float a1[256];
  __shared__ float a2[128];
  const float* xr = last + b * 512;
  const float* w1 = hw1 + (long long)hh * 512 * 256;
  for (int j = threadIdx.x; j < 256; j += blockDim.x) {
    float acc = hb1[hh * 256 + j];
    for (int kk = 0; kk < 512; ++kk) acc += xr[kk] * w1[kk * 256 + j];
    a1[j] = gelu_t(acc);
  }
  __syncthreads();
  const float* w2 = hw2 + (long long)hh * 256 * 128;
  for (int j = threadIdx.x; j < 128; j += blockDim.x) {
    float acc = hb2[hh * 128 + j];
    for (int kk = 0; kk < 256; ++kk) acc += a1[kk] * w2[kk * 128 + j];
    a2[j] = gelu_t(acc);
  }
  __syncthreads();
  if (threadIdx.x < 32) {
    const float* w3 = hw3 + hh * 128;
    float acc = 0.f;
    for (int kk = threadIdx.x; kk < 128; kk += 32) acc += a2[kk] * w3[kk];
    #pragma unroll
    for (int off = 16; off; off >>= 1) acc += __shfl_xor(acc, off);
    if (threadIdx.x == 0) out[b * 6 + hh] = acc + hb3[hh];
  }
}

// ---------------------------------------------------------------- host orchestration
extern "C" void kernel_launch(void* const* d_in, const int* in_sizes, int n_in,
                              void* d_out, int out_size, void* d_ws, size_t ws_size,
                              hipStream_t stream)
{
  (void)in_sizes; (void)n_in; (void)out_size; (void)ws_size;
  const float* X       = (const float*)d_in[0];
  const float* w_in    = (const float*)d_in[1];
  const float* b_in    = (const float*)d_in[2];
  const float* ln_in_s = (const float*)d_in[3];
  const float* ln_in_b = (const float*)d_in[4];
  const float* s_ln_s  = (const float*)d_in[5];
  const float* s_ln_b  = (const float*)d_in[6];
  const float* s_conv_w= (const float*)d_in[7];
  const float* s_conv_b= (const float*)d_in[8];
  const float* s_wg    = (const float*)d_in[9];
  const float* s_rg    = (const float*)d_in[10];
  const float* s_bg    = (const float*)d_in[11];
  const float* s_gn    = (const float*)d_in[12];
  const float* s_ffln_s= (const float*)d_in[13];
  const float* s_ffln_b= (const float*)d_in[14];
  const float* s_ff_w1 = (const float*)d_in[15];
  const float* s_ff_w2 = (const float*)d_in[16];
  const float* m_ln_s  = (const float*)d_in[17];
  const float* m_ln_b  = (const float*)d_in[18];
  const float* m_w_up  = (const float*)d_in[19];
  const float* m_conv_w= (const float*)d_in[20];
  const float* m_conv_b= (const float*)d_in[21];
  const float* m_wq    = (const float*)d_in[22];
  const float* m_wk    = (const float*)d_in[23];
  const float* m_wv    = (const float*)d_in[24];
  const float* m_w_if  = (const float*)d_in[25];
  const float* m_b_if  = (const float*)d_in[26];
  const float* m_skip  = (const float*)d_in[27];
  const float* m_gn    = (const float*)d_in[28];
  const float* m_w_down= (const float*)d_in[29];
  const float* post_s  = (const float*)d_in[30];
  const float* post_b  = (const float*)d_in[31];
  const float* hw1 = (const float*)d_in[32];
  const float* hb1 = (const float*)d_in[33];
  const float* hw2 = (const float*)d_in[34];
  const float* hb2 = (const float*)d_in[35];
  const float* hw3 = (const float*)d_in[36];
  const float* hb3 = (const float*)d_in[37];

  const int B = 32, T = 252, D = 512, NH = 8, DI = 1024, FF = 682;
  const int FFP = 688;                               // padded ld for act buffer (16B rows)
  const long long BT = (long long)B * T;             // 8064
  const int BTi = (int)BT;

  float* ws = (float*)d_ws;
  float* hbuf  = ws;  ws += BT * D;
  float* lnbuf = ws;  ws += BT * D;
  float* big   = ws;  ws += BT * 2 * DI;             // ud (mLSTM) / u (FF)
  float* xc    = ws;  ws += BT * DI;
  float* g0    = ws;  ws += BT * DI;                 // pre_i / q
  float* g1    = ws;  ws += BT * DI;                 // pre_f / k
  float* g2    = ws;  ws += BT * DI;                 // pre_z / v
  float* g3    = ws;  ws += BT * DI;                 // pre_o / attn
  float* hsbuf = ws;  ws += BT * D;
  float* actb  = ws;  ws += BT * FFP;
  float* gates = ws;  ws += BT * 16;
  float* cfb   = ws;  ws += (long long)B * NH * T;
  float* lib   = ws;  ws += (long long)B * NH * T;
  float* fused = ws;  ws += BT * DI;
  __bf16* rtb  = (__bf16*)ws;  ws += (4 * 8 * 64 * 64) / 2;   // bf16 transposed R

  auto gemm = [&](const float* A, const float* W, const float* bias, float* C,
                  int M, int N, int K, int lda, int ldw, int ldc,
                  long long aB, long long wB, long long cB, int batch,
                  float alpha, int accum) {
    dim3 grid((N + BN - 1) / BN, (M + BM - 1) / BM, batch);
    k_gemm<<<grid, 128, 0, stream>>>(A, W, bias, C, M, N, K, lda, ldw, ldc,
                                     aB, wB, cB, alpha, accum);
  };

  k_input_proj<<<BTi, 256, 0, stream>>>(X, w_in, b_in, ln_in_s, ln_in_b, hbuf);

  int si = 0, mi = 0;
  for (int blk = 0; blk < 7; ++blk) {
    if ((blk & 1) == 0) {
      // ---------------- sLSTM block ----------------
      k_layernorm<<<BTi, 256, 0, stream>>>(hbuf, s_ln_s + si * D, s_ln_b + si * D, lnbuf, D, D);
      {
        long long tot = BT * D;
        k_conv_silu<<<(unsigned)((tot + 255) / 256), 256, 0, stream>>>(
            lnbuf, s_conv_w + (long long)si * 4 * D, s_conv_b + (long long)si * D,
            xc, T, D, D, tot);
      }
      const float* Wg = s_wg + (long long)si * 4 * D * D;
      const float* bg = s_bg + (long long)si * 4 * D;
      gemm(xc,    Wg + 0LL * D * D, bg + 0 * D, g0, BTi, D, D, D, D, D, 0, 0, 0, 1, 1.f, 0);
      gemm(xc,    Wg + 1LL * D * D, bg + 1 * D, g1, BTi, D, D, D, D, D, 0, 0, 0, 1, 1.f, 0);
      gemm(lnbuf, Wg + 2LL * D * D, bg + 2 * D, g2, BTi, D, D, D, D, D, 0, 0, 0, 1, 1.f, 0);
      gemm(lnbuf, Wg + 3LL * D * D, bg + 3 * D, g3, BTi, D, D, D, D, D, 0, 0, 0, 1, 1.f, 0);
      k_prep_rt<<<(4 * 8 * 64 * 64 + 255) / 256, 256, 0, stream>>>(
          s_rg + (long long)si * 4 * NH * 64 * 64, rtb);
      k_slstm_scan<<<2, 512, 0, stream>>>(g0, g1, g2, g3, rtb, hsbuf, T);
      k_headnorm_add<<<BTi, 256, 0, stream>>>(hsbuf, s_gn + si * D, hbuf, D, 64);
      // FF
      k_layernorm<<<BTi, 256, 0, stream>>>(hbuf, s_ffln_s + si * D, s_ffln_b + si * D, lnbuf, D, D);
      gemm(lnbuf, s_ff_w1 + (long long)si * D * 1364, nullptr, big,
           BTi, 1364, D, D, 1364, 1364, 0, 0, 0, 1, 1.f, 0);
      {
        long long tot = BT * FF;
        k_gelu_mul<<<(unsigned)((tot + 255) / 256), 256, 0, stream>>>(big, actb, FF, 1364, FFP, tot);
      }
      gemm(actb, s_ff_w2 + (long long)si * FF * D, nullptr, hbuf,
           BTi, D, FF, FFP, D, D, 0, 0, 0, 1, 1.f, 1);
      ++si;
    } else {
      // ---------------- mLSTM block ----------------
      k_layernorm<<<BTi, 256, 0, stream>>>(hbuf, m_ln_s + mi * D, m_ln_b + mi * D, lnbuf, D, D);
      gemm(lnbuf, m_w_up + (long long)mi * D * 2 * DI, nullptr, big,
           BTi, 2 * DI, D, D, 2 * DI, 2 * DI, 0, 0, 0, 1, 1.f, 0);
      {
        long long tot = BT * DI;   // xc = silu(conv(xm)); xm = big[:, :DI], ld = 2*DI
        k_conv_silu<<<(unsigned)((tot + 255) / 256), 256, 0, stream>>>(
            big, m_conv_w + (long long)mi * 4 * DI, m_conv_b + (long long)mi * DI,
            xc, T, DI, 2 * DI, tot);
      }
      const long long wqB = 128LL * 128;
      gemm(xc,  m_wq + (long long)mi * NH * wqB, nullptr, g0,
           BTi, 128, 128, DI, 128, DI, 128, wqB, 128, NH, 1.f, 0);
      gemm(xc,  m_wk + (long long)mi * NH * wqB, nullptr, g1,
           BTi, 128, 128, DI, 128, DI, 128, wqB, 128, NH, 0.0883883476483184f, 0); // 1/sqrt(128)
      gemm(big, m_wv + (long long)mi * NH * wqB, nullptr, g2,
           BTi, 128, 128, 2 * DI, 128, DI, 128, wqB, 128, NH, 1.f, 0);  // v from xm
      {
        long long tot = BT * 16;
        k_gates<<<(unsigned)((tot + 255) / 256), 256, 0, stream>>>(
            xc, m_w_if + (long long)mi * DI * 16, m_b_if + mi * 16, gates, DI, tot);
      }
      k_gatescan<<<1, 256, 0, stream>>>(gates, cfb, lib, T);
      {
        dim3 ag((T + 15) / 16, NH, B);
        k_mlstm_attn<<<ag, 32, 0, stream>>>(g0, g1, g2, cfb, lib, g3, T, DI);
      }
      k_mlstm_fuse<<<BTi, 256, 0, stream>>>(g3, m_gn + mi * DI, m_skip + mi * DI,
                                            xc, big, fused, DI, 128, 2 * DI);
      gemm(fused, m_w_down + (long long)mi * DI * D, nullptr, hbuf,
           BTi, D, DI, DI, D, D, 0, 0, 0, 1, 1.f, 1);
      ++mi;
    }
  }
  // post-LN on last token of every batch row: rows at h + (T-1)*D, stride T*D
  k_layernorm<<<B, 256, 0, stream>>>(hbuf + (long long)(T - 1) * D, post_s, post_b,
                                     lnbuf, D, (long long)T * D);
  k_heads<<<dim3(6, B), 128, 0, stream>>>(lnbuf, hw1, hb1, hw2, hb2, hw3, hb3, (float*)d_out);
}
